// CLS_5669356833410
// MI455X (gfx1250) — matrix operand order repaired and checked by reading, not compiled, not run on hardware
//
#include <hip/hip_runtime.h>
#include <hip/hip_bf16.h>

// ---------------------------------------------------------------------------
// CDNA5 (gfx1250) implementation.
//   - 1x1 convs (reduce 64->16, expand 16->64) as WMMA GEMMs
//   - fft2/ifft2 over 298x298 (padded to 320) as symmetric DFT-GEMMs:
//       P(X) = (X*F)^T,  P(P(X)) = F*X*F = fft2(X)   (F symmetric)
//     so B is ALWAYS the twiddle matrix: staged to LDS once per block via
//     global_load_async_to_lds_b128 (ASYNCcnt) and read with ds_load_b128;
//     A fragments / C stores are contiguous b128 vector ops.
//   - OTFs of the 21x21 blur kernel / 3x3 kernel_P evaluated per-frequency
//   - 3x3 convs / pooling / tiny 1x1 on scalar VALU (≈3.5 GFLOP, negligible)
// ---------------------------------------------------------------------------

typedef __attribute__((ext_vector_type(16))) _Float16 v16h;
typedef __attribute__((ext_vector_type(8)))  _Float16 v8h;
typedef __attribute__((ext_vector_type(8)))  float    v8f;
typedef __attribute__((ext_vector_type(4)))  float    v4f;

#define NP   320            // padded FFT dim (>=298, multiple of 32)
#define NP2  (NP * NP)      // 102400
#define HP   298            // true FFT dim
#define PI2  6.28318530717958647692f
#define SB_LD 328           // LDS row stride (f16) for twiddle panel (pad vs banks)
#define USE_ASYNC_LDS 1     // global_load_async_to_lds_b128 staging

__device__ __forceinline__ v8f wmma16(v16h a, v16h b, v8f c) {
  return __builtin_amdgcn_wmma_f32_16x16x32_f16(
      /*neg_a=*/false, a, /*neg_b=*/false, b,
      /*c_mod=*/(short)0, c, /*reuse_a=*/false, /*reuse_b=*/false);
}

__device__ __forceinline__ unsigned lds_off(const void* p) {
  return (unsigned)(size_t)(const __attribute__((address_space(3))) char*)p;
}

__device__ __forceinline__ v16h cat8(v8h lo, v8h hi) {
  v16h r;
#pragma unroll
  for (int i = 0; i < 8; ++i) { r[i] = lo[i]; r[8 + i] = hi[i]; }
  return r;
}

// A fragment (16x32 f16 tile, row-major, stride NP): per lane two contiguous
// 16B chunks: k in [16h,16h+8) and [16h+8,16h+16)  -> global_load_b128 x2
__device__ __forceinline__ v16h load_a_vec(const _Float16* A, int lane) {
  int row = lane & 15, half = lane >> 4;
  const v8h* p0 = (const v8h*)(A + row * NP + 16 * half);
  const v8h* p1 = (const v8h*)(A + row * NP + 16 * half + 8);
  return cat8(*p0, *p1);
}

// B fragment from LDS twiddle panel (symmetric matrix read row-wise):
// lane reads contiguous k in [16h + 32ks, +16)  -> ds_load_b128 x2
__device__ __forceinline__ v16h load_b_lds(const _Float16* sB, int ks, int lane) {
  int col = lane & 15, half = lane >> 4;
  const v8h* q = (const v8h*)(sB + col * SB_LD + ks * 32 + 16 * half);
  return cat8(q[0], q[1]);
}

// ---------------------------------------------------------------------------
// Kernel 1: reduce 1x1 conv: cls[b][16][HW] = W[16][64] * x[b][64][HW] + bias
// ---------------------------------------------------------------------------
__global__ void k_reduce(const float* __restrict__ x, const float* __restrict__ w,
                         const float* __restrict__ bias, float* __restrict__ cls) {
  int wave = blockIdx.x * (blockDim.x >> 5) + (threadIdx.x >> 5);
  int lane = threadIdx.x & 31;
  if (wave >= 4 * 4096) return;
  int b    = wave >> 12;
  int pos0 = (wave & 4095) * 16;
  const float* xb = x + (long)b * 64 * 65536;
  int half = lane >> 4, li = lane & 15;
  v8f acc = {};
#pragma unroll
  for (int ks = 0; ks < 2; ++ks) {
    v16h af, bf;
#pragma unroll
    for (int v = 0; v < 8; ++v) {
      int kb = 2 * v + half * 16 + ks * 32;
      af[2 * v]     = (_Float16)w[li * 64 + kb];
      af[2 * v + 1] = (_Float16)w[li * 64 + kb + 1];
    }
#pragma unroll
    for (int v = 0; v < 8; ++v) {
      int k = 2 * v + half * 16 + ks * 32;
      bf[2 * v]     = (_Float16)xb[(long)k * 65536 + pos0 + li];
      bf[2 * v + 1] = (_Float16)xb[(long)(k + 1) * 65536 + pos0 + li];
    }
    acc = wmma16(af, bf, acc);
  }
#pragma unroll
  for (int r = 0; r < 8; ++r) {
    int m = r + 8 * half;
    cls[(long)b * 16 * 65536 + (long)m * 65536 + pos0 + li] = acc[r] + bias[m];
  }
}

// ---------------------------------------------------------------------------
// 3x3 VALID conv, C=16->16, optional leaky-relu (slope=1 -> identity).
// ---------------------------------------------------------------------------
__global__ void k_conv3(const float* __restrict__ in, int Hin,
                        const float* __restrict__ w, const float* __restrict__ bias,
                        float* __restrict__ out, int Hout, float slope) {
  long idx   = (long)blockIdx.x * blockDim.x + threadIdx.x;
  long total = 4L * 16 * Hout * Hout;
  if (idx >= total) return;
  int xo = (int)(idx % Hout);
  long t = idx / Hout;
  int yo = (int)(t % Hout); t /= Hout;
  int co = (int)(t % 16);
  int b  = (int)(t / 16);
  const float* ib = in + (long)b * 16 * Hin * Hin;
  float acc = bias[co];
  for (int ci = 0; ci < 16; ++ci) {
    const float* ip = ib + (long)ci * Hin * Hin + (long)yo * Hin + xo;
    const float* wp = w + (co * 16 + ci) * 9;
#pragma unroll
    for (int ky = 0; ky < 3; ++ky)
#pragma unroll
      for (int kx = 0; kx < 3; ++kx)
        acc = fmaf(ip[ky * Hin + kx], wp[ky * 3 + kx], acc);
  }
  out[idx] = (acc < 0.f) ? acc * slope : acc;
}

// Adaptive 3x3 average pool on (4,16,Hg,Hg) -> (4,16,3,3)
__global__ void k_pool(const float* __restrict__ g, int Hg, float* __restrict__ pooled) {
  int idx = blockIdx.x * blockDim.x + threadIdx.x;
  if (idx >= 576) return;
  int j = idx % 3, i = (idx / 3) % 3, c = (idx / 9) % 16, b = idx / 144;
  int r0 = i * Hg / 3, r1 = ((i + 1) * Hg + 2) / 3;
  int c0 = j * Hg / 3, c1 = ((j + 1) * Hg + 2) / 3;
  const float* p = g + ((long)b * 16 + c) * Hg * Hg;
  float s = 0.f;
  for (int r = r0; r < r1; ++r)
    for (int cc = c0; cc < c1; ++cc) s += p[r * Hg + cc];
  pooled[(b * 16 + c) * 9 + i * 3 + j] = s / (float)((r1 - r0) * (c1 - c0));
}

// 1x1 conv on 3x3 maps + exp + subtract spatial mean -> kernel_P
__global__ void k_g4(const float* __restrict__ pooled, const float* __restrict__ w4,
                     const float* __restrict__ b4, float* __restrict__ kp) {
  __shared__ float vals[144];
  int b = blockIdx.x, t = threadIdx.x;
  if (t < 144) {
    int c = t / 9, s = t % 9;
    float acc = b4[c];
    for (int ci = 0; ci < 16; ++ci)
      acc = fmaf(w4[c * 16 + ci], pooled[(b * 16 + ci) * 9 + s], acc);
    vals[t] = expf(acc);
  }
  __syncthreads();
  if (t < 16) {
    float m = 0.f;
    for (int s = 0; s < 9; ++s) m += vals[t * 9 + s];
    m *= (1.f / 9.f);
    for (int s = 0; s < 9; ++s) kp[(b * 16 + t) * 9 + s] = vals[t * 9 + s] - m;
  }
}

// Edge-pad cls (256x256) by 21 -> f16 320x320 (zero outside 298x298)
__global__ void k_pad(const float* __restrict__ cls, _Float16* __restrict__ X) {
  long idx = (long)blockIdx.x * blockDim.x + threadIdx.x;
  if (idx >= 64L * NP2) return;
  int x  = (int)(idx % NP);
  int y  = (int)((idx / NP) % NP);
  int bc = (int)(idx / NP2);
  float v = 0.f;
  if (x < HP && y < HP) {
    int yi = min(max(y - 21, 0), 255);
    int xi = min(max(x - 21, 0), 255);
    v = cls[(long)bc * 65536 + yi * 256 + xi];
  }
  X[idx] = (_Float16)v;
}

// Forward + inverse DFT twiddle matrices (f16, 320x320, zero beyond 298)
__global__ void k_twiddle(_Float16* Fre, _Float16* Fim, _Float16* Gre, _Float16* Gim) {
  int idx = blockIdx.x * blockDim.x + threadIdx.x;
  if (idx >= NP2) return;
  int k = idx % NP, c = idx / NP;
  float re = 0.f, im = 0.f, gre = 0.f, gim = 0.f;
  if (c < HP && k < HP) {
    long m = ((long)c * k) % HP;
    float ang = -PI2 * (float)m / (float)HP;
    re  = __cosf(ang);
    im  = __sinf(ang);
    gre = re * (1.f / (float)HP);     // conj / Hp
    gim = -im * (1.f / (float)HP);
  }
  Fre[idx] = (_Float16)re;  Fim[idx] = (_Float16)im;
  Gre[idx] = (_Float16)gre; Gim[idx] = (_Float16)gim;
}

// Kf = psf2otf(kernel 21x21 rolled by (-10,-10)) evaluated directly per freq.
__global__ void k_Kf(const float* __restrict__ ker, float* __restrict__ Kre,
                     float* __restrict__ Kim) {
  long idx = (long)blockIdx.x * blockDim.x + threadIdx.x;
  if (idx >= 4L * NP2) return;
  int v = (int)(idx % NP);
  int u = (int)((idx / NP) % NP);
  int b = (int)(idx / NP2);
  float re = 0.f, im = 0.f;
  if (u < HP && v < HP) {
    const float* kb = ker + b * 441;
    for (int i = 0; i < 21; ++i)
      for (int j = 0; j < 21; ++j) {
        long p = (long)u * (i - 10) + (long)v * (j - 10);
        p = ((p % HP) + HP) % HP;
        float ang = -PI2 * (float)p / (float)HP;
        float kv  = kb[i * 21 + j];
        re = fmaf(kv, __cosf(ang), re);
        im = fmaf(kv, __sinf(ang), im);
      }
  }
  Kre[idx] = re; Kim[idx] = im;
}

// prod = conj(Kf)/(|Kf|^2+|Pf|^2) * Z ; Pf (3x3, roll -1,-1) computed on the fly
__global__ void k_point(const float* __restrict__ Kre, const float* __restrict__ Kim,
                        const float* __restrict__ kp,
                        const _Float16* __restrict__ Zre, const _Float16* __restrict__ Zim,
                        _Float16* __restrict__ Pre, _Float16* __restrict__ Pim) {
  long idx = (long)blockIdx.x * blockDim.x + threadIdx.x;
  if (idx >= 64L * NP2) return;
  int v  = (int)(idx % NP);
  int u  = (int)((idx / NP) % NP);
  int bc = (int)(idx / NP2);
  int b  = bc >> 4;
  float outr = 0.f, outi = 0.f;
  if (u < HP && v < HP) {
    float pr = 0.f, pi = 0.f;
    const float* kpp = kp + bc * 9;
    for (int i = 0; i < 3; ++i)
      for (int j = 0; j < 3; ++j) {
        long p = (long)u * (i - 1) + (long)v * (j - 1);
        p = ((p % HP) + HP) % HP;
        float ang = -PI2 * (float)p / (float)HP;
        float pv  = kpp[i * 3 + j];
        pr = fmaf(pv, __cosf(ang), pr);
        pi = fmaf(pv, __sinf(ang), pi);
      }
    float kr = Kre[(long)b * NP2 + u * NP + v];
    float ki = Kim[(long)b * NP2 + u * NP + v];
    float denom = kr * kr + ki * ki + pr * pr + pi * pi;
    float inv = 1.f / denom;
    float ir = kr * inv, ii = -ki * inv;
    float zr = (float)Zre[idx], zi = (float)Zim[idx];
    outr = ir * zr - ii * zi;
    outi = ir * zi + ii * zr;
  }
  Pre[idx] = (_Float16)outr;
  Pim[idx] = (_Float16)outi;
}

// ---------------------------------------------------------------------------
// Symmetric DFT-GEMM pass:  C = (A * F)^T      (F = 320x320 symmetric twiddle)
//   A: batched MxK f16 (re/im; A_REAL => imag==0), row-major stride NP
//   F: shared complex twiddle; staged once per block into LDS (async copy)
//   C: transposed store -> per-lane contiguous b128 stores
//   OUT_REAL: store only Re(C) as f32 (final IFFT pass)
// Block = 128 threads = 4 waves sharing one nt (16 twiddle rows), mt = mg*4+w.
// ---------------------------------------------------------------------------
template <bool A_REAL, bool OUT_REAL>
__global__ void k_cgemm2(const _Float16* __restrict__ Are, const _Float16* __restrict__ Aim,
                         long aBatch,
                         const _Float16* __restrict__ Fre, const _Float16* __restrict__ Fim,
                         _Float16* __restrict__ Cre, _Float16* __restrict__ Cim,
                         float* __restrict__ Cf32, int nBatch) {
  __shared__ _Float16 sBre[16 * SB_LD + 8];
  __shared__ _Float16 sBim[16 * SB_LD + 8];

  int blk  = blockIdx.x;
  int mg   = blk % 5;
  int nt   = (blk / 5) % 20;
  int bidx = blk / 100;
  if (bidx >= nBatch) return;
  int w    = threadIdx.x >> 5;
  int lane = threadIdx.x & 31;
  int mt   = mg * 4 + w;

  // ---- stage the 16-row twiddle panel (full K=320, re+im) into LDS ----
  {
    int t = threadIdx.x;
#pragma unroll
    for (int i = 0; i < 5; ++i) {
      int lin = i * 128 + t;              // 0..639 ; 40 x 16B chunks per row
      int r = lin / 40, c = lin % 40;
      const _Float16* gre = Fre + (long)(nt * 16 + r) * NP + c * 8;
      const _Float16* gim = Fim + (long)(nt * 16 + r) * NP + c * 8;
      _Float16* lre = &sBre[r * SB_LD + c * 8];
      _Float16* lim = &sBim[r * SB_LD + c * 8];
#if USE_ASYNC_LDS
      asm volatile("global_load_async_to_lds_b128 %0, %1, off"
                   :: "v"(lds_off(lre)), "v"((unsigned long long)(size_t)gre)
                   : "memory");
      asm volatile("global_load_async_to_lds_b128 %0, %1, off"
                   :: "v"(lds_off(lim)), "v"((unsigned long long)(size_t)gim)
                   : "memory");
#else
      *(v8h*)lre = *(const v8h*)gre;
      *(v8h*)lim = *(const v8h*)gim;
#endif
    }
#if USE_ASYNC_LDS
    asm volatile("s_wait_asynccnt 0x0" ::: "memory");
#endif
    __syncthreads();
  }

  const _Float16* aRe = Are + (long)bidx * aBatch + (long)mt * 16 * NP;
  const _Float16* aIm = A_REAL ? nullptr : (Aim + (long)bidx * aBatch + (long)mt * 16 * NP);

  v8f accR = {};
  v8f accI = {};
#pragma unroll 2
  for (int ks = 0; ks < NP / 32; ++ks) {
    if (ks + 1 < NP / 32) __builtin_prefetch(aRe + (ks + 1) * 32, 0, 1);
    v16h ar = load_a_vec(aRe + ks * 32, lane);
    v16h br = load_b_lds(sBre, ks, lane);
    v16h bi = load_b_lds(sBim, ks, lane);
    accR = wmma16(ar, br, accR);                  // += Are*Fre
    if (!OUT_REAL) accI = wmma16(ar, bi, accI);   // += Are*Fim
    if (!A_REAL) {
      v16h ai  = load_a_vec(aIm + ks * 32, lane);
      v16h nai = -ai;                             // f16 WMMA lacks A-negate: VALU negate
      accR = wmma16(nai, bi, accR);               // -= Aim*Fim
      if (!OUT_REAL) accI = wmma16(ai, br, accI); // += Aim*Fre
    }
  }

  // ---- transposed store: C^T[gn][mt*16 + 8h .. +8) contiguous per lane ----
  int half = lane >> 4, col = lane & 15;
  int gn = nt * 16 + col;
  long rowOff = (long)bidx * NP2 + (long)gn * NP + mt * 16 + 8 * half;
  if (OUT_REAL) {
    v4f lo = {accR[0], accR[1], accR[2], accR[3]};
    v4f hi = {accR[4], accR[5], accR[6], accR[7]};
    *(v4f*)(Cf32 + rowOff)     = lo;
    *(v4f*)(Cf32 + rowOff + 4) = hi;
  } else {
    v8h pr, pi;
#pragma unroll
    for (int r = 0; r < 8; ++r) { pr[r] = (_Float16)accR[r]; pi[r] = (_Float16)accI[r]; }
    *(v8h*)(Cre + rowOff) = pr;
    *(v8h*)(Cim + rowOff) = pi;
  }
}

// ---------------------------------------------------------------------------
// Expand 1x1 conv: out[b][64][HW] = Wexp[64][16] * clear[16][HW] + bias
// ---------------------------------------------------------------------------
__global__ void k_expand(const float* __restrict__ deblur, const float* __restrict__ w,
                         const float* __restrict__ bias, float* __restrict__ out) {
  int wave = blockIdx.x * (blockDim.x >> 5) + (threadIdx.x >> 5);
  int lane = threadIdx.x & 31;
  if (wave >= 4 * 4 * 4096) return;
  int b    = wave / (4 * 4096);
  int rem  = wave % (4 * 4096);
  int mt   = rem / 4096;
  int pos0 = (rem % 4096) * 16;
  int half = lane >> 4, li = lane & 15;

  v16h af, bf;
#pragma unroll
  for (int v = 0; v < 8; ++v) {
    int kb = 2 * v + half * 16;
    int m  = mt * 16 + li;
    af[2 * v]     = (kb < 16)     ? (_Float16)w[m * 16 + kb]     : (_Float16)0.f;
    af[2 * v + 1] = (kb + 1 < 16) ? (_Float16)w[m * 16 + kb + 1] : (_Float16)0.f;
  }
  int pos = pos0 + li;
  int y = pos >> 8, x = pos & 255;
#pragma unroll
  for (int v = 0; v < 8; ++v) {
    int k = 2 * v + half * 16;
    bf[2 * v] = (k < 16)
        ? (_Float16)deblur[((long)b * 16 + k) * NP2 + (long)(y + 21) * NP + (x + 21)]
        : (_Float16)0.f;
    bf[2 * v + 1] = (k + 1 < 16)
        ? (_Float16)deblur[((long)b * 16 + k + 1) * NP2 + (long)(y + 21) * NP + (x + 21)]
        : (_Float16)0.f;
  }
  v8f acc = {};
  acc = wmma16(af, bf, acc);
#pragma unroll
  for (int r = 0; r < 8; ++r) {
    int m = mt * 16 + r + 8 * half;
    out[((long)b * 64 + m) * 65536 + pos0 + li] = acc[r] + bias[m];
  }
}

// ---------------------------------------------------------------------------
extern "C" void kernel_launch(void* const* d_in, const int* in_sizes, int n_in,
                              void* d_out, int out_size, void* d_ws, size_t ws_size,
                              hipStream_t stream) {
  const float* x        = (const float*)d_in[0];
  const float* kernel   = (const float*)d_in[1];
  const float* w_reduce = (const float*)d_in[2];
  const float* b_reduce = (const float*)d_in[3];
  const float* w_g1     = (const float*)d_in[4];
  const float* b_g1     = (const float*)d_in[5];
  const float* w_g2     = (const float*)d_in[6];
  const float* b_g2     = (const float*)d_in[7];
  const float* w_g3     = (const float*)d_in[8];
  const float* b_g3     = (const float*)d_in[9];
  const float* w_g4     = (const float*)d_in[10];
  const float* b_g4     = (const float*)d_in[11];
  const float* w_expand = (const float*)d_in[12];
  const float* b_expand = (const float*)d_in[13];
  float* out = (float*)d_out;

  char* p = (char*)d_ws;
  auto alloc = [&](size_t bytes) -> void* {
    void* r = (void*)p;
    p += (bytes + 255) & ~(size_t)255;
    return r;
  };
  const long ACS = 64L * NP2;
  float*     cls    = (float*)alloc(4L * 16 * 65536 * 4);
  float*     g1     = (float*)alloc(4L * 16 * 254 * 254 * 4);
  float*     g2     = (float*)alloc(4L * 16 * 252 * 252 * 4);
  float*     g3     = (float*)alloc(4L * 16 * 250 * 250 * 4);
  float*     pooled = (float*)alloc(576 * 4);
  float*     kp     = (float*)alloc(576 * 4);
  _Float16*  Xpad   = (_Float16*)alloc(ACS * 2);
  _Float16*  FwRe   = (_Float16*)alloc(NP2 * 2);
  _Float16*  FwIm   = (_Float16*)alloc(NP2 * 2);
  _Float16*  FiRe   = (_Float16*)alloc(NP2 * 2);
  _Float16*  FiIm   = (_Float16*)alloc(NP2 * 2);
  _Float16*  YrRe   = (_Float16*)alloc(ACS * 2);
  _Float16*  YrIm   = (_Float16*)alloc(ACS * 2);
  _Float16*  ZRe    = (_Float16*)alloc(ACS * 2);
  _Float16*  ZIm    = (_Float16*)alloc(ACS * 2);
  _Float16*  PrRe   = (_Float16*)alloc(ACS * 2);
  _Float16*  PrIm   = (_Float16*)alloc(ACS * 2);
  _Float16*  TRe    = (_Float16*)alloc(ACS * 2);
  _Float16*  TIm    = (_Float16*)alloc(ACS * 2);
  float*     KfRe   = (float*)alloc(4L * NP2 * 4);
  float*     KfIm   = (float*)alloc(4L * NP2 * 4);
  float*     deblur = (float*)alloc(ACS * 4);
  (void)ws_size; (void)n_in; (void)in_sizes; (void)out_size;

  // twiddles
  k_twiddle<<<(NP2 + 255) / 256, 256, 0, stream>>>(FwRe, FwIm, FiRe, FiIm);

  // reduce 1x1 conv (WMMA)
  k_reduce<<<4096, 128, 0, stream>>>(x, w_reduce, b_reduce, cls);

  // 3x3 conv chain
  {
    long t1 = 4L * 16 * 254 * 254;
    k_conv3<<<(unsigned)((t1 + 255) / 256), 256, 0, stream>>>(cls, 256, w_g1, b_g1, g1, 254, 0.1f);
    long t2 = 4L * 16 * 252 * 252;
    k_conv3<<<(unsigned)((t2 + 255) / 256), 256, 0, stream>>>(g1, 254, w_g2, b_g2, g2, 252, 0.1f);
    long t3 = 4L * 16 * 250 * 250;
    k_conv3<<<(unsigned)((t3 + 255) / 256), 256, 0, stream>>>(g2, 252, w_g3, b_g3, g3, 250, 1.0f);
  }

  // pool + kernel_P
  k_pool<<<3, 256, 0, stream>>>(g3, 250, pooled);
  k_g4<<<4, 256, 0, stream>>>(pooled, w_g4, b_g4, kp);

  // edge pad -> f16
  k_pad<<<(unsigned)((64L * NP2 + 255) / 256), 256, 0, stream>>>(cls, Xpad);

  // blur-kernel OTF
  k_Kf<<<(unsigned)((4L * NP2 + 255) / 256), 256, 0, stream>>>(kernel, KfRe, KfIm);

  // forward fft2 = P(P(X)), P(X) = (X*Fw)^T
  const int gemmBlocks = 64 * 20 * 5;   // batch * nt * mt-groups ; 128 thr/block
  k_cgemm2<true, false><<<gemmBlocks, 128, 0, stream>>>(
      Xpad, nullptr, (long)NP2, FwRe, FwIm, YrRe, YrIm, nullptr, 64);
  k_cgemm2<false, false><<<gemmBlocks, 128, 0, stream>>>(
      YrRe, YrIm, (long)NP2, FwRe, FwIm, ZRe, ZIm, nullptr, 64);

  // Wiener pointwise
  k_point<<<(unsigned)((64L * NP2 + 255) / 256), 256, 0, stream>>>(
      KfRe, KfIm, kp, ZRe, ZIm, PrRe, PrIm);

  // inverse ifft2 = P'(P'(Prod)), P'(X) = (X*Fi)^T ; final pass real f32
  k_cgemm2<false, false><<<gemmBlocks, 128, 0, stream>>>(
      PrRe, PrIm, (long)NP2, FiRe, FiIm, TRe, TIm, nullptr, 64);
  k_cgemm2<false, true><<<gemmBlocks, 128, 0, stream>>>(
      TRe, TIm, (long)NP2, FiRe, FiIm, nullptr, nullptr, deblur, 64);

  // expand 1x1 conv (WMMA)
  k_expand<<<16384, 128, 0, stream>>>(deblur, w_expand, b_expand, out);
}
